// FMOperatorLayer_13408887898822
// MI455X (gfx1250) — matrix operand order repaired and checked by reading, not compiled, not run on hardware
//
#include <hip/hip_runtime.h>
#include <hip/hip_bf16.h>
#include <math.h>

// ---------------- constants ----------------
#define BATCH_  4
#define LEN_    2048
#define DIM_    768
#define INNER_  2304
#define NFFT_   4096
#define FORD_   64
#define BL_     (BATCH_ * LEN_)     // 8192 GEMM rows

typedef __attribute__((ext_vector_type(16))) __bf16 bf16x16;
typedef __attribute__((ext_vector_type(8)))  __bf16 bf16x8;
typedef __attribute__((ext_vector_type(8)))  float  f32x8;

__device__ __constant__ const float kPI = 3.14159265358979323846f;

__device__ inline float2 cmul(float2 a, float2 b) {
  return make_float2(a.x * b.x - a.y * b.y, a.x * b.y + a.y * b.x);
}

// ---------------- fp32 -> bf16 convert ----------------
__global__ void cvt_bf16_kernel(const float* __restrict__ src, __bf16* __restrict__ dst, long n) {
  long i = (long)blockIdx.x * blockDim.x + threadIdx.x;
  long stride = (long)gridDim.x * blockDim.x;
  for (; i < n; i += stride) dst[i] = (__bf16)src[i];
}

// ---------------- bf16 WMMA GEMM:  C[m][n] = sum_k A[m][k] * W[n][k] + bias[n] ----------------
// Block tile 64(M) x 128(N); 8 waves/block, each wave owns a 64x16 strip (4 WMMA
// accumulators) so every B fragment is reused 4x. A tile (64 x 768 bf16 = 96 KB)
// staged into dynamic LDS with GLOBAL_LOAD_ASYNC_TO_LDS_B128 (ASYNCcnt path).
// K must be 768 (true for both GEMMs here).
__global__ void gemm_bf16_wmma_kernel(const __bf16* __restrict__ A,
                                      const __bf16* __restrict__ W,
                                      const float*  __restrict__ bias,
                                      float* __restrict__ C,
                                      int M, int N, int K) {
  extern __shared__ __bf16 As[];           // [64][768]
  const int wave = threadIdx.x >> 5;       // 0..7
  const int lane = threadIdx.x & 31;
  const int tilesN = N / 128;
  const int m0 = (blockIdx.x / tilesN) * 64;
  const int n0 = (blockIdx.x % tilesN) * 128 + wave * 16;

  // ---- async-stage the 64 x K A-tile into LDS: 6144 chunks of 16 B ----
  {
    const int chunksPerRow = (K * 2) / 16;             // 96
    for (int c = threadIdx.x; c < 64 * chunksPerRow; c += 256) {
      int row  = c / chunksPerRow;
      int colb = (c - row * chunksPerRow) * 16;        // byte offset in row
      const char* g = (const char*)(A + (size_t)(m0 + row) * K) + colb;
      unsigned lds = (unsigned)(uintptr_t)((char*)As + (size_t)c * 16);
      asm volatile("global_load_async_to_lds_b128 %0, %1, off"
                   :: "v"(lds), "v"(g) : "memory");
    }
    asm volatile("s_wait_asynccnt 0x0" ::: "memory");
  }
  __syncthreads();

  const int mrow  = lane & 15;     // A row within 16-tile / B,N column
  const int khalf = lane >> 4;     // 0 or 1

  f32x8 acc[4] = {};
  for (int kk = 0; kk < K; kk += 32) {
    // ---- B fragment: B[k][n] = W[n][k]; lanes 0-15 K=kk..kk+7,kk+16..kk+23, lanes 16-31 K=kk+8..kk+15,kk+24..kk+31
    const __bf16* bp = W + (size_t)(n0 + mrow) * 768 + kk + khalf * 8;
    __builtin_prefetch(bp + 32, 0, 1);
    bf16x8 bLo = *(const bf16x8*)bp;
    bf16x8 bHi = *(const bf16x8*)(bp + 16);
    bf16x16 b;
#pragma unroll
    for (int j = 0; j < 8; ++j) { b[j] = bLo[j]; b[j + 8] = bHi[j]; }

    // ---- 4 A fragments (16-bit A 16x32 CDNA5 lane layout), 4 WMMAs per B load
#pragma unroll
    for (int t = 0; t < 4; ++t) {
      const __bf16* ap = As + (size_t)(t * 16 + mrow) * 768 + kk;
      bf16x8 aLo = *(const bf16x8*)(ap + khalf * 8);
      bf16x8 aHi = *(const bf16x8*)(ap + 16 + khalf * 8);
      bf16x16 a;
#pragma unroll
      for (int j = 0; j < 8; ++j) { a[j] = aLo[j]; a[j + 8] = aHi[j]; }
      acc[t] = __builtin_amdgcn_wmma_f32_16x16x32_bf16(false, a, false, b, (short)0,
                                                       acc[t], false, false);
    }
  }

  // ---- D: VGPR r -> lanes 0-15 M=r, lanes 16-31 M=r+8; N = lane&15
  const float bs = bias[n0 + mrow];
#pragma unroll
  for (int t = 0; t < 4; ++t) {
#pragma unroll
    for (int r = 0; r < 8; ++r) {
      int row = t * 16 + r + khalf * 8;
      C[(size_t)(m0 + row) * N + (n0 + mrow)] = acc[t][r] + bs;
    }
  }
}

// ---------------- GELU + depthwise conv3 + split + v*x1 ----------------
// u: [B][L][INNER] ; outputs gate[B][D][L], vprod[B][D][L]
__global__ void conv_split_kernel(const float* __restrict__ u,
                                  const float* __restrict__ convw,
                                  const float* __restrict__ convb,
                                  float* __restrict__ gate,
                                  float* __restrict__ vprod) {
  long i = (long)blockIdx.x * blockDim.x + threadIdx.x;
  const long total = (long)BATCH_ * DIM_ * LEN_;
  if (i >= total) return;
  int l = (int)(i % LEN_);
  int d = (int)((i / LEN_) % DIM_);
  int b = (int)(i / ((long)DIM_ * LEN_));

  auto conv_at = [&](int c) -> float {
    float s = convb[c];
#pragma unroll
    for (int j = 0; j < 3; ++j) {
      int ll = l - 1 + j;
      if (ll >= 0 && ll < LEN_) {
        float x = u[((size_t)b * LEN_ + ll) * INNER_ + c];
        float ge = 0.5f * x * (1.0f + erff(x * 0.70710678118654752f));  // exact GELU
        s += convw[c * 3 + j] * ge;
      }
    }
    return s;
  };

  size_t o = ((size_t)b * DIM_ + d) * LEN_ + l;
  gate[o]  = conv_at(d);
  vprod[o] = conv_at(2 * DIM_ + d) * conv_at(DIM_ + d);
}

// ---------------- implicit SIREN filter -> kfil[D][L] ----------------
__global__ void filter_kernel(const float* __restrict__ posz,
                              const float* __restrict__ w0, const float* __restrict__ b0,
                              const float* __restrict__ sfreq,
                              const float* __restrict__ w1, const float* __restrict__ b1,
                              const float* __restrict__ w2, const float* __restrict__ b2,
                              const float* __restrict__ wout,
                              float* __restrict__ kfil) {
  int l = blockIdx.x * blockDim.x + threadIdx.x;
  if (l >= LEN_) return;
  float z0 = posz[l * 3 + 0], z1 = posz[l * 3 + 1], z2 = posz[l * 3 + 2];
  float ha[FORD_], hb[FORD_];
  for (int i = 0; i < FORD_; ++i)
    ha[i] = sinf(sfreq[i] * (z0 * w0[i * 3] + z1 * w0[i * 3 + 1] + z2 * w0[i * 3 + 2] + b0[i]));
  for (int i = 0; i < FORD_; ++i) {
    float s = b1[i];
    for (int j = 0; j < FORD_; ++j) s += ha[j] * w1[i * FORD_ + j];
    hb[i] = sinf(sfreq[i] * s);
  }
  for (int i = 0; i < FORD_; ++i) {
    float s = b2[i];
    for (int j = 0; j < FORD_; ++j) s += hb[j] * w2[i * FORD_ + j];
    ha[i] = sinf(sfreq[i] * s);
  }
  const float tt = (float)l / (float)(LEN_ - 1);
  const float max_decay = logf(0.02f) / 0.3f;
  const float min_decay = logf(0.02f) / 1.5f;
  for (int d = 0; d < DIM_; ++d) {
    float s = 0.f;
    for (int j = 0; j < FORD_; ++j) s += ha[j] * wout[d * FORD_ + j];
    float delta = min_decay + (max_decay - min_decay) * ((float)d / (float)(DIM_ - 1));
    kfil[(size_t)d * LEN_ + l] = s * expf(-tt * fabsf(delta));
  }
}

// ---------------- alpha = sigmoid(alpha_param) ----------------
__global__ void alpha_kernel(const float* __restrict__ ap, float* __restrict__ alph) {
  int i = blockIdx.x * blockDim.x + threadIdx.x;
  if (i < DIM_) alph[i] = 1.0f / (1.0f + expf(-ap[i]));
}

// ---------------- in-place radix-2 FFT of 4096 complex in LDS (256 threads) ----------------
__device__ inline void fft4096(float2* s, int tid, float dir) {
  __syncthreads();
  for (int i = tid; i < NFFT_; i += 256) {
    unsigned j = __brev((unsigned)i) >> 20;  // 12-bit reversal
    if (j > (unsigned)i) { float2 t = s[i]; s[i] = s[j]; s[j] = t; }
  }
  __syncthreads();
  for (int len = 2; len <= NFFT_; len <<= 1) {
    int half = len >> 1;
    float ang0 = dir * 2.0f * kPI / (float)len;
    for (int k = tid; k < NFFT_ / 2; k += 256) {
      int blk = k / half;
      int pos = k - blk * half;
      int i0 = blk * len + pos;
      int i1 = i0 + half;
      float sn, cs;
      __sincosf(ang0 * (float)pos, &sn, &cs);
      float2 a = s[i0], bb = s[i1];
      float2 tb = cmul(bb, make_float2(cs, sn));
      s[i0] = make_float2(a.x + tb.x, a.y + tb.y);
      s[i1] = make_float2(a.x - tb.x, a.y - tb.y);
    }
    __syncthreads();
  }
}

// ---------------- precompute chirp spectrum Gf[d][4096] for sign*alpha ----------------
__global__ void chirp_fft_kernel(const float* __restrict__ alph, float sign,
                                 float2* __restrict__ Gf) {
  __shared__ float2 s[NFFT_];
  int d = blockIdx.x, tid = threadIdx.x;
  float alpha = sign * alph[d];
  float phi = alpha * (kPI * 0.5f);
  float sp = sinf(phi);
  float sl = (fabsf(sp) < 1e-8f) ? 1.0f : sp;
  float csc = 1.0f / sl;
  for (int m = tid; m < NFFT_; m += 256) {
    float n = (float)m - (float)LEN_;
    float ss, cc;
    __sincosf(kPI * csc * n * n, &ss, &cc);
    s[m] = make_float2(cc, ss);
  }
  fft4096(s, tid, -1.0f);
  for (int m = tid; m < NFFT_; m += 256) Gf[(size_t)d * NFFT_ + m] = s[m];
}

// ---------------- fractional FT of one column (Bluestein via precomputed Gf) ----------------
// in_r != null : real input column (vprod / kfil)
// else         : complex input = in_c[col] * in_c2[d]   (v_frft * k_frft)
// out_c != null: complex output; else real-part output to out_r
__global__ void frft_kernel(const float* __restrict__ in_r,
                            const float2* __restrict__ in_c,
                            const float2* __restrict__ in_c2,
                            const float* __restrict__ alph, float sign,
                            const float2* __restrict__ Gf,
                            float2* __restrict__ out_c,
                            float* __restrict__ out_r) {
  __shared__ float2 s[NFFT_];
  const int col = blockIdx.x, tid = threadIdx.x;
  const int d = col % DIM_;
  const float alpha = sign * alph[d];
  const float phi = alpha * (kPI * 0.5f);
  const float sp = sinf(phi), cp = cosf(phi);
  const float sl = (fabsf(sp) < 1e-8f) ? 1.0f : sp;
  const float cot = cp / sl;

  // load + pre-chirp, zero-pad
  for (int l = tid; l < LEN_; l += 256) {
    float t = (float)l - (float)(LEN_ - 1) * 0.5f;
    float ss, cc;
    __sincosf(-kPI * cot * t * t, &ss, &cc);
    float2 pre = make_float2(cc, ss);
    float2 x;
    if (in_r) {
      float xr = in_r[(size_t)col * LEN_ + l];
      x = make_float2(xr * cc, xr * ss);
    } else {
      float2 p = cmul(in_c[(size_t)col * LEN_ + l], in_c2[(size_t)d * LEN_ + l]);
      x = cmul(p, pre);
    }
    s[l] = x;
  }
  for (int l = LEN_ + tid; l < NFFT_; l += 256) s[l] = make_float2(0.f, 0.f);

  fft4096(s, tid, -1.0f);                        // forward FFT
  for (int m = tid; m < NFFT_; m += 256)         // x chirp spectrum
    s[m] = cmul(s[m], Gf[(size_t)d * NFFT_ + m]);
  fft4096(s, tid, +1.0f);                        // inverse FFT (scale below)

  float sr, si;
  __sincosf(-(1.0f - alpha) * (kPI * 0.25f), &si, &sr);
  float mag = 1.0f / sqrtf(fabsf(sl));
  float2 scale = make_float2(sr * mag, si * mag);
  const float invN = 1.0f / (float)NFFT_;

  for (int l = tid; l < LEN_; l += 256) {
    float t = (float)l - (float)(LEN_ - 1) * 0.5f;
    float ss, cc;
    __sincosf(-kPI * cot * t * t, &ss, &cc);
    float2 pre = make_float2(cc, ss);
    float2 c = s[l + LEN_ - 1];
    c.x *= invN; c.y *= invN;
    float2 o = cmul(cmul(scale, pre), c);
    if (out_c) out_c[(size_t)col * LEN_ + l] = o;
    else       out_r[(size_t)col * LEN_ + l] = o.x;
  }
}

// ---------------- head norm + head mix + gate -> bf16 rows for final GEMM ----------------
__global__ void norm_gate_kernel(const float* __restrict__ v2,    // [B][D][L]
                                 const float* __restrict__ gate,  // [B][D][L]
                                 const float* __restrict__ nscale,  // [192]
                                 const float* __restrict__ nparams, // [4][4]
                                 __bf16* __restrict__ yb) {        // [B*L][D]
  const int bl = blockIdx.x;
  const int b = bl >> 11;            // /2048
  const int l = bl & (LEN_ - 1);
  const int tid = threadIdx.x;
  __shared__ float ssum[4], ssq[4], sstd[4];
  __shared__ float sv[DIM_];
  if (tid < 4) { ssum[tid] = 0.f; ssq[tid] = 0.f; }
  __syncthreads();

#pragma unroll
  for (int r = 0; r < 3; ++r) {
    int dch = tid + r * 256;
    float v = v2[((size_t)b * DIM_ + dch) * LEN_ + l];
    int h = dch / 192;
    atomicAdd(&ssum[h], v);
    atomicAdd(&ssq[h], v * v);
    sv[dch] = v;  // raw for now, normalize after std known
  }
  __syncthreads();
  if (tid < 4) {
    float mean = ssum[tid] / 192.0f;
    float var = (ssq[tid] - 192.0f * mean * mean) / 191.0f;
    sstd[tid] = sqrtf(fmaxf(var, 0.0f)) + 1e-5f;
  }
  __syncthreads();
#pragma unroll
  for (int r = 0; r < 3; ++r) {
    int dch = tid + r * 256;
    int h = dch / 192, dd = dch - h * 192;
    sv[dch] = nscale[dd] * sv[dch] / sstd[h];   // normalized
  }
  __syncthreads();
#pragma unroll
  for (int r = 0; r < 3; ++r) {
    int dch = tid + r * 256;
    int h = dch / 192, dd = dch - h * 192;
    float y = 0.f;
#pragma unroll
    for (int hp = 0; hp < 4; ++hp)
      y += sv[hp * 192 + dd] * nparams[hp * 4 + h];
    float g = gate[((size_t)b * DIM_ + dch) * LEN_ + l];
    y *= 1.0f / (1.0f + expf(-g));
    yb[((size_t)b * LEN_ + l) * DIM_ + dch] = (__bf16)y;
  }
}

// ---------------- launcher ----------------
extern "C" void kernel_launch(void* const* d_in, const int* in_sizes, int n_in,
                              void* d_out, int out_size, void* d_ws, size_t ws_size,
                              hipStream_t stream) {
  const float* x       = (const float*)d_in[0];
  const float* ipw     = (const float*)d_in[1];
  const float* ipb     = (const float*)d_in[2];
  const float* convw   = (const float*)d_in[3];
  const float* convb   = (const float*)d_in[4];
  const float* alphap  = (const float*)d_in[5];
  const float* posz    = (const float*)d_in[6];
  const float* w0      = (const float*)d_in[7];
  const float* b0      = (const float*)d_in[8];
  const float* sfreq   = (const float*)d_in[9];
  const float* w1      = (const float*)d_in[10];
  const float* b1      = (const float*)d_in[11];
  const float* w2      = (const float*)d_in[12];
  const float* b2      = (const float*)d_in[13];
  const float* wout    = (const float*)d_in[14];
  const float* nscale  = (const float*)d_in[15];
  const float* nparams = (const float*)d_in[16];
  const float* opw     = (const float*)d_in[17];
  const float* opb     = (const float*)d_in[18];
  float* out = (float*)d_out;

  char* ws = (char*)d_ws;
  size_t off = 0;
  auto alloc = [&](size_t bytes) -> char* {
    off = (off + 255) & ~(size_t)255;
    char* p = ws + off;
    off += bytes;
    return p;
  };

  __bf16* xb   = (__bf16*)alloc((size_t)BL_ * DIM_ * 2);
  __bf16* wb   = (__bf16*)alloc((size_t)INNER_ * DIM_ * 2);
  __bf16* wob  = (__bf16*)alloc((size_t)DIM_ * DIM_ * 2);
  float*  u    = (float*)alloc((size_t)BL_ * INNER_ * 4);
  float*  gate = (float*)alloc((size_t)BATCH_ * DIM_ * LEN_ * 4);
  float*  vpr  = (float*)alloc((size_t)BATCH_ * DIM_ * LEN_ * 4);
  float*  kfil = (float*)alloc((size_t)DIM_ * LEN_ * 4);
  float*  alph = (float*)alloc((size_t)DIM_ * 4);
  float2* Gfp  = (float2*)alloc((size_t)DIM_ * NFFT_ * 8);
  float2* Gfm  = (float2*)alloc((size_t)DIM_ * NFFT_ * 8);
  float2* vf   = (float2*)alloc((size_t)BATCH_ * DIM_ * LEN_ * 8);
  float2* kf   = (float2*)alloc((size_t)DIM_ * LEN_ * 8);
  float*  v2   = (float*)alloc((size_t)BATCH_ * DIM_ * LEN_ * 4);
  __bf16* yb   = (__bf16*)alloc((size_t)BL_ * DIM_ * 2);

  // 1) operand conversion to bf16
  cvt_bf16_kernel<<<256, 256, 0, stream>>>(x,   xb,  (long)BL_ * DIM_);
  cvt_bf16_kernel<<<256, 256, 0, stream>>>(ipw, wb,  (long)INNER_ * DIM_);
  cvt_bf16_kernel<<<256, 256, 0, stream>>>(opw, wob, (long)DIM_ * DIM_);

  const size_t ldsA = (size_t)64 * 768 * sizeof(__bf16);   // 96 KB dynamic LDS

  // 2) in_proj GEMM (WMMA bf16, f32 accumulate)
  gemm_bf16_wmma_kernel<<<(BL_ / 64) * (INNER_ / 128), 256, ldsA, stream>>>(
      xb, wb, ipb, u, BL_, INNER_, DIM_);

  // 3) GELU + depthwise conv3 + split + v*x1
  conv_split_kernel<<<(BATCH_ * DIM_ * LEN_) / 256, 256, 0, stream>>>(
      u, convw, convb, gate, vpr);

  // 4) implicit filter + alpha
  filter_kernel<<<LEN_ / 256, 256, 0, stream>>>(posz, w0, b0, sfreq, w1, b1, w2, b2, wout, kfil);
  alpha_kernel<<<(DIM_ + 255) / 256, 256, 0, stream>>>(alphap, alph);

  // 5) chirp spectra for +alpha and -alpha
  chirp_fft_kernel<<<DIM_, 256, 0, stream>>>(alph, +1.0f, Gfp);
  chirp_fft_kernel<<<DIM_, 256, 0, stream>>>(alph, -1.0f, Gfm);

  // 6) forward frft of v and k; inverse frft of product (real part)
  frft_kernel<<<BATCH_ * DIM_, 256, 0, stream>>>(vpr,  nullptr, nullptr, alph, +1.0f, Gfp, vf, nullptr);
  frft_kernel<<<DIM_,          256, 0, stream>>>(kfil, nullptr, nullptr, alph, +1.0f, Gfp, kf, nullptr);
  frft_kernel<<<BATCH_ * DIM_, 256, 0, stream>>>(nullptr, vf, kf, alph, -1.0f, Gfm, nullptr, v2);

  // 7) head norm + mix + gate -> bf16
  norm_gate_kernel<<<BL_, 256, 0, stream>>>(v2, gate, nscale, nparams, yb);

  // 8) out_proj GEMM (WMMA bf16)
  gemm_bf16_wmma_kernel<<<(BL_ / 64) * (DIM_ / 128), 256, ldsA, stream>>>(
      yb, wob, opb, out, BL_, DIM_, DIM_);
}
